// Attend_39986145526500
// MI455X (gfx1250) — matrix operand order repaired and checked by reading, not compiled, not run on hardware
//
#include <hip/hip_runtime.h>

// ---- CDNA5 (gfx1250) sliding-window flash attention ----
// f16 WMMA / f32 accum, double-buffered ASYNC global->LDS staging.
// B=4 H=16 S=4096 D=64 W=256.  One workgroup per (b*h, t) query block.
// 8 waves (wave32) x 32 query rows each; keys streamed in chunks of 32.

typedef _Float16 f16;
typedef __attribute__((ext_vector_type(16))) _Float16    v16h;
typedef __attribute__((ext_vector_type(8)))  float       v8f;
typedef __attribute__((ext_vector_type(4)))  unsigned int u32x4;
typedef __attribute__((ext_vector_type(4)))  float       f32x4;

#define SEQ   4096
#define DIM   64
#define WIN   256
#define TBLK  (SEQ / WIN)   // 16 query blocks per (b,h)
#define KC    32            // key chunk (one WMMA k-step for P*V)
#define NCH   16            // 2W / KC key chunks per query block
#define KSTR  72            // Ks LDS row stride (f16 elems): 144B -> conflict-free
#define VSTR  40            // Vt LDS row stride: 80B -> conflict-free
#define PSTR  40            // Ps LDS row stride

union Frag {
  v16h  v;
  u32x4 q[2];
};

// Build a 16-element f16 fragment from two 16-byte LDS runs.
__device__ __forceinline__ v16h ld_frag2(const f16* p0, const f16* p1) {
  Frag f;
  f.q[0] = *(const u32x4*)p0;
  f.q[1] = *(const u32x4*)p1;
  return f.v;
}

// Async-copy this thread's 32B of K and V for chunk c into LDS staging.
// gfx1250: GLOBAL_LOAD_ASYNC_TO_LDS_B128 (ASYNCcnt), inline asm for portability.
__device__ __forceinline__ void issue_async_chunk(const float* __restrict__ K,
                                                  const float* __restrict__ V,
                                                  long long kvbase, int c, int tid,
                                                  float* KrawBuf, float* VrawBuf) {
  const int li = tid * 8;                 // 8 floats per thread per matrix
  const int kk = li >> 6;                 // key within chunk
  const int dd = li & 63;                 // d offset (multiple of 8)
  const long long r = kvbase + (long long)c * KC + kk;
  const unsigned long long gk =
      (unsigned long long)(uintptr_t)(K + r * DIM + dd);
  const unsigned long long gv =
      (unsigned long long)(uintptr_t)(V + r * DIM + dd);
  const unsigned lk = (unsigned)(uintptr_t)(KrawBuf + li);
  const unsigned lv = (unsigned)(uintptr_t)(VrawBuf + li);
  asm volatile("global_load_async_to_lds_b128 %0, %1, off"
               :: "v"(lk), "v"(gk) : "memory");
  asm volatile("global_load_async_to_lds_b128 %0, %1, off offset:16"
               :: "v"(lk), "v"(gk) : "memory");
  asm volatile("global_load_async_to_lds_b128 %0, %1, off"
               :: "v"(lv), "v"(gv) : "memory");
  asm volatile("global_load_async_to_lds_b128 %0, %1, off offset:16"
               :: "v"(lv), "v"(gv) : "memory");
}

__global__ __launch_bounds__(256, 1)
void swa_fwd_kernel(const float* __restrict__ Q, const float* __restrict__ K,
                    const float* __restrict__ V, float* __restrict__ O) {
  __shared__ float Kraw[2][KC * DIM];     // raw f32 staging (ping-pong), 2x8KB
  __shared__ float Vraw[2][KC * DIM];     // 2x8KB
  __shared__ f16   Ks[KC * KSTR];         // K chunk, row-major [key][d]
  __shared__ f16   Vt[DIM * VSTR];        // V chunk, transposed [d][key]
  __shared__ f16   Ps[8][32 * PSTR];      // per-wave P tile [row][key]

  const int blk  = blockIdx.x;
  const int t    = blk & (TBLK - 1);
  const int bh   = blk >> 4;
  const int tid  = threadIdx.x;
  const int w    = tid >> 5;      // wave id 0..7
  const int L    = tid & 31;      // lane
  const int col  = L & 15;
  const int half = L >> 4;

  const long long qrow0 = (long long)bh * SEQ + (long long)t * WIN;

  // ---- Q A-fragments (f16), kept in registers for the whole tile ----
  // A 16-bit layout: lane holds row=col, K = half*8+{0..7} and 16+half*8+{0..7}
  v16h aq[2][2];
#pragma unroll
  for (int mt = 0; mt < 2; ++mt) {
#pragma unroll
    for (int ks = 0; ks < 2; ++ks) {
      const float* qp = Q + (qrow0 + w * 32 + mt * 16 + col) * DIM
                          + ks * 32 + half * 8;
      f32x4 a0 = *(const f32x4*)(qp);
      f32x4 a1 = *(const f32x4*)(qp + 4);
      f32x4 a2 = *(const f32x4*)(qp + 16);
      f32x4 a3 = *(const f32x4*)(qp + 20);
      Frag f;
      f16* h = (f16*)&f;
#pragma unroll
      for (int j = 0; j < 4; ++j) {
        h[j]      = (f16)a0[j];
        h[4 + j]  = (f16)a1[j];
        h[8 + j]  = (f16)a2[j];
        h[12 + j] = (f16)a3[j];
      }
      aq[mt][ks] = f.v;
    }
  }

  // ---- online-softmax state (C-layout rows: row = mt*16 + half*8 + g) ----
  float mrow[2][8], lrow[2][8];
  v8f   oacc[2][4];
#pragma unroll
  for (int mt = 0; mt < 2; ++mt)
#pragma unroll
    for (int g = 0; g < 8; ++g) { mrow[mt][g] = -3.0e38f; lrow[mt][g] = 0.0f; }
#pragma unroll
  for (int mt = 0; mt < 2; ++mt)
#pragma unroll
    for (int nt = 0; nt < 4; ++nt)
#pragma unroll
      for (int g = 0; g < 8; ++g) oacc[mt][nt][g] = 0.0f;

  const int c0 = (t == 0) ? 8 : 0;                         // j >= 0 for t==0
  const long long kvbase = (long long)bh * SEQ + (long long)t * WIN - WIN;

  // prologue: kick off async staging of the first chunk
  issue_async_chunk(K, V, kvbase, c0, tid, Kraw[c0 & 1], Vraw[c0 & 1]);

  for (int c = c0; c < NCH; ++c) {
    // this wave's async writes for chunk c are resident after this wait
    asm volatile("s_wait_asynccnt 0x0" ::: "memory");
    __syncthreads();   // all waves resident + previous chunk fully consumed

    // ---- convert own raw f32 region -> f16 Ks / Vt (8 elems each) ----
    {
      const int li = tid * 8;
      const int kk = li >> 6;
      const int dd = li & 63;
      const int buf = c & 1;
      f32x4 k0 = *(const f32x4*)&Kraw[buf][li];
      f32x4 k1 = *(const f32x4*)&Kraw[buf][li + 4];
      f32x4 v0 = *(const f32x4*)&Vraw[buf][li];
      f32x4 v1 = *(const f32x4*)&Vraw[buf][li + 4];
      union { f16 h[8]; u32x4 q; } pk;
#pragma unroll
      for (int j = 0; j < 4; ++j) { pk.h[j] = (f16)k0[j]; pk.h[4 + j] = (f16)k1[j]; }
      *(u32x4*)&Ks[kk * KSTR + dd] = pk.q;
#pragma unroll
      for (int j = 0; j < 4; ++j) {
        Vt[(dd + j) * VSTR + kk]     = (f16)v0[j];
        Vt[(dd + 4 + j) * VSTR + kk] = (f16)v1[j];
      }
    }
    // overlap: start async staging of next chunk into the other buffer
    if (c + 1 < NCH)
      issue_async_chunk(K, V, kvbase, c + 1, tid, Kraw[(c + 1) & 1], Vraw[(c + 1) & 1]);
    __syncthreads();

    // wave-uniform band pruning: wave w needs kloc in [w*32, w*32+287]
    if (c >= w && c <= w + 8) {
      const bool edge = (c == w) || (c == w + 8);  // only edges need the mask

      // ---- B fragments for Q*K^T (B 16-bit: K = half*8+{0..7} and 16+half*8+{0..7}) ----
      v16h bk[2][2];
#pragma unroll
      for (int nt = 0; nt < 2; ++nt)
#pragma unroll
        for (int ks = 0; ks < 2; ++ks) {
          const f16* p = &Ks[(nt * 16 + col) * KSTR + ks * 32 + half * 8];
          bk[nt][ks] = ld_frag2(p, p + 16);
        }

#pragma unroll
      for (int mt = 0; mt < 2; ++mt) {
        // S = Q*K^T for this m-tile (two 16x16 n-tiles, K-dim 64 = 2 steps)
        v8f sv[2];
#pragma unroll
        for (int nt = 0; nt < 2; ++nt) {
          v8f acc;
#pragma unroll
          for (int g = 0; g < 8; ++g) acc[g] = 0.0f;
          acc = __builtin_amdgcn_wmma_f32_16x16x32_f16(
              false, aq[mt][0], false, bk[nt][0], (short)0, acc, false, false);
          acc = __builtin_amdgcn_wmma_f32_16x16x32_f16(
              false, aq[mt][1], false, bk[nt][1], (short)0, acc, false, false);
          sv[nt] = acc;
        }

        // ---- scale (+ band mask only on edge chunks) ----
        float sm[2][8];
        const int qb = w * 32 + mt * 16 + half * 8;
#pragma unroll
        for (int nt = 0; nt < 2; ++nt)
#pragma unroll
          for (int g = 0; g < 8; ++g) sm[nt][g] = sv[nt][g] * 0.125f;  // d^-0.5
        if (edge) {
#pragma unroll
          for (int nt = 0; nt < 2; ++nt) {
            const int kloc = c * KC + nt * 16 + col;
#pragma unroll
            for (int g = 0; g < 8; ++g) {
              const int qloc = qb + g;
              const bool ok = (kloc >= qloc) && (kloc <= qloc + WIN);
              sm[nt][g] = ok ? sm[nt][g] : -3.0e38f;
            }
          }
        }

        // ---- online softmax (row reduce over 16 lanes of the half-group) --
        float alpha[8];
#pragma unroll
        for (int g = 0; g < 8; ++g) {
          float rm = fmaxf(sm[0][g], sm[1][g]);
#pragma unroll
          for (int off = 1; off < 16; off <<= 1)
            rm = fmaxf(rm, __shfl_xor(rm, off, 32));
          const float mn = fmaxf(mrow[mt][g], rm);
          alpha[g] = __expf(mrow[mt][g] - mn);
          mrow[mt][g] = mn;
          const float p0 = __expf(sm[0][g] - mn);
          const float p1 = __expf(sm[1][g] - mn);
          sm[0][g] = p0; sm[1][g] = p1;
          float rs = p0 + p1;
#pragma unroll
          for (int off = 1; off < 16; off <<= 1)
            rs += __shfl_xor(rs, off, 32);
          lrow[mt][g] = lrow[mt][g] * alpha[g] + rs;
        }
#pragma unroll
        for (int nt = 0; nt < 4; ++nt)
#pragma unroll
          for (int g = 0; g < 8; ++g) oacc[mt][nt][g] *= alpha[g];

        // ---- stage P (C-layout -> LDS row-major, read back in A-layout) ---
#pragma unroll
        for (int nt = 0; nt < 2; ++nt)
#pragma unroll
          for (int g = 0; g < 8; ++g) {
            const int r = mt * 16 + half * 8 + g;
            Ps[w][r * PSTR + nt * 16 + col] = (f16)sm[nt][g];
          }
      }

      // ---- O += P * V (K-dim = 32 keys = one WMMA per out tile) ----
#pragma unroll
      for (int mt = 0; mt < 2; ++mt) {
        const f16* pa = &Ps[w][(mt * 16 + col) * PSTR + half * 8];
        const v16h ap = ld_frag2(pa, pa + 16);
#pragma unroll
        for (int nt = 0; nt < 4; ++nt) {
          const f16* pb = &Vt[(nt * 16 + col) * VSTR + half * 8];
          const v16h bv = ld_frag2(pb, pb + 16);
          oacc[mt][nt] = __builtin_amdgcn_wmma_f32_16x16x32_f16(
              false, ap, false, bv, (short)0, oacc[mt][nt], false, false);
        }
      }
    }
  }

  // ---- epilogue: O = acc / l ----
#pragma unroll
  for (int mt = 0; mt < 2; ++mt)
#pragma unroll
    for (int g = 0; g < 8; ++g) {
      const float inv = 1.0f / lrow[mt][g];
      const long long row = qrow0 + w * 32 + mt * 16 + half * 8 + g;
      float* op = O + row * DIM + col;
#pragma unroll
      for (int nt = 0; nt < 4; ++nt) op[nt * 16] = oacc[mt][nt][g] * inv;
    }
}

extern "C" void kernel_launch(void* const* d_in, const int* in_sizes, int n_in,
                              void* d_out, int out_size, void* d_ws, size_t ws_size,
                              hipStream_t stream) {
  (void)in_sizes; (void)n_in; (void)out_size; (void)d_ws; (void)ws_size;
  const float* q = (const float*)d_in[0];
  const float* k = (const float*)d_in[1];
  const float* v = (const float*)d_in[2];
  float* out = (float*)d_out;
  // grid = B*H*T = 4*16*16 = 1024 workgroups of 256 threads (8 wave32)
  swa_fwd_kernel<<<dim3(4 * 16 * TBLK), dim3(256), 0, stream>>>(q, k, v, out);
}
